// MambaWrapper_56556129354034
// MI455X (gfx1250) — matrix operand order repaired and checked by reading, not compiled, not run on hardware
//
#include <hip/hip_runtime.h>
#include <hip/hip_bf16.h>

// ---------------- problem constants ----------------
#define BB      2
#define LL      2048
#define DMODEL  512
#define DINNER  1024
#define DSTATE  16
#define DCONV   4
#define DTRANK  32
#define MROWS   (BB * LL)          // 4096 flattened (b,l) rows
#define XPROJ   (DTRANK + 2*DSTATE) // 64

typedef unsigned short ushort_t;

typedef __attribute__((ext_vector_type(16))) __bf16 bf16x16;
typedef __attribute__((ext_vector_type(8)))  __bf16 bf16x8;
typedef __attribute__((ext_vector_type(8)))  float  f32x8;

union BF16Frag { bf16x16 v; bf16x8 h[2]; };

__device__ __forceinline__ ushort_t f32_to_bf16(float f) {
    unsigned int u = __float_as_uint(f);
    u = (u + 0x7FFFu + ((u >> 16) & 1u)) >> 16;   // round-nearest-even
    return (ushort_t)u;
}
__device__ __forceinline__ float bf16_to_f32(ushort_t h) {
    return __uint_as_float(((unsigned int)h) << 16);
}

// ---------------- fp32 -> bf16 conversion ----------------
__global__ void cvt_f32_bf16_kernel(const float* __restrict__ in,
                                    ushort_t* __restrict__ out, int n) {
    int i = blockIdx.x * 256 + threadIdx.x;
    if (i < n) out[i] = f32_to_bf16(in[i]);
}

// ---------------- generic WMMA bf16 GEMM: C[M,N] = A[M,K] * W[N,K]^T ----------------
// one wave per block; wave computes a 16(M) x 64(N) strip. K % 32 == 0, N % 64 == 0.
__global__ __launch_bounds__(32)
void gemm_bf16_wmma_kernel(const ushort_t* __restrict__ A, int lda,
                           const ushort_t* __restrict__ W, int ldw,
                           float* __restrict__ C, int ldc,
                           int K, int betaAdd) {
    const int lane  = threadIdx.x;          // 0..31
    const int mTile = blockIdx.x;           // M/16 tiles
    const int n0    = blockIdx.y * 64;
    const int sub   = lane & 15;
    const int hiA   = (lane & 16) ? 16 : 0; // A K-base, contiguous K order matching B
    const int hiB   = (lane & 16) ? 16 : 0; // B K-base, contiguous K order matching A

    const ushort_t* Arow = A + (size_t)(mTile * 16 + sub) * lda;
    const ushort_t* W0 = W + (size_t)(n0 +  0 + sub) * ldw;
    const ushort_t* W1 = W + (size_t)(n0 + 16 + sub) * ldw;
    const ushort_t* W2 = W + (size_t)(n0 + 32 + sub) * ldw;
    const ushort_t* W3 = W + (size_t)(n0 + 48 + sub) * ldw;

    f32x8 acc0 = {}, acc1 = {}, acc2 = {}, acc3 = {};

    for (int k0 = 0; k0 < K; k0 += 32) {
        BF16Frag a;
        a.h[0] = *(const bf16x8*)(Arow + k0 + hiA);        // K = base .. base+7
        a.h[1] = *(const bf16x8*)(Arow + k0 + hiA + 8);    // K = base+8 .. base+15
        const int kb = k0 + hiB;
        bf16x16 b0 = *(const bf16x16*)(W0 + kb);
        bf16x16 b1 = *(const bf16x16*)(W1 + kb);
        bf16x16 b2 = *(const bf16x16*)(W2 + kb);
        bf16x16 b3 = *(const bf16x16*)(W3 + kb);
        acc0 = __builtin_amdgcn_wmma_f32_16x16x32_bf16(false, a.v, false, b0, (short)0, acc0, false, false);
        acc1 = __builtin_amdgcn_wmma_f32_16x16x32_bf16(false, a.v, false, b1, (short)0, acc1, false, false);
        acc2 = __builtin_amdgcn_wmma_f32_16x16x32_bf16(false, a.v, false, b2, (short)0, acc2, false, false);
        acc3 = __builtin_amdgcn_wmma_f32_16x16x32_bf16(false, a.v, false, b3, (short)0, acc3, false, false);
    }

    // C/D layout: VGPR v holds row (v + 8*(lane>=16)), col = lane&15
    const int rbase = mTile * 16 + ((lane >> 4) << 3);
    float* Cp = C + (size_t)rbase * ldc + n0 + sub;
#pragma unroll
    for (int v = 0; v < 8; ++v) {
        float* cr = Cp + (size_t)v * ldc;
        if (betaAdd) {
            cr[0]  += acc0[v]; cr[16] += acc1[v]; cr[32] += acc2[v]; cr[48] += acc3[v];
        } else {
            cr[0]  = acc0[v]; cr[16] = acc1[v]; cr[32] = acc2[v]; cr[48] = acc3[v];
        }
    }
}

// ---------------- depthwise causal conv (width 4) + bias + SiLU -> bf16 ----------------
// dir=0: taps t-3..t (forward).  dir=1: taps t..t+3 (reverse direction in original coords)
__global__ void conv_silu_kernel(const float* __restrict__ xz,     // [B,L,2*DINNER], xs = ch 0..DINNER-1
                                 const float* __restrict__ convw,  // [DINNER,1,4]
                                 const float* __restrict__ convb,  // [DINNER]
                                 ushort_t* __restrict__ xsb,       // [B,L,DINNER] bf16
                                 int dir) {
    int i = blockIdx.x * 256 + threadIdx.x;
    if (i >= BB * LL * DINNER) return;
    int d = i % DINNER;
    int t = (i / DINNER) % LL;
    int b = i / (DINNER * LL);
    float s = 0.f;
#pragma unroll
    for (int j = 0; j < DCONV; ++j) {
        int tt = dir ? (t + (DCONV - 1) - j) : (t - (DCONV - 1) + j);
        float v = (tt >= 0 && tt < LL)
                  ? xz[((size_t)(b * LL + tt)) * (2 * DINNER) + d] : 0.f;
        s += convw[d * DCONV + j] * v;
    }
    s += convb[d];
    float sig = 1.f / (1.f + __expf(-s));
    xsb[i] = f32_to_bf16(s * sig);
}

// ---------------- dt = softplus(raw + dtb), in-place ----------------
__global__ void softplus_kernel(float* __restrict__ dt, const float* __restrict__ dtb, int n) {
    int i = blockIdx.x * 256 + threadIdx.x;
    if (i >= n) return;
    float x = dt[i] + dtb[i % DINNER];
    dt[i] = (x > 20.f) ? x : __logf(1.f + __expf(x));
}

// ---------------- fused selective scan + gating epilogue ----------------
// one lane per (b,d) channel; 16-state recurrence held in registers;
// shared B/C vectors staged through LDS once per timestep.
__global__ __launch_bounds__(256)
void scan_kernel(const float*  __restrict__ dt,    // [B,L,DINNER]
                 const ushort_t* __restrict__ xsb, // [B,L,DINNER] bf16
                 const float*  __restrict__ xdbl,  // [B,L,64]: B at 32..47, C at 48..63
                 const float*  __restrict__ Alog,  // [DINNER,16]
                 const float*  __restrict__ Dp,    // [DINNER]
                 const float*  __restrict__ xz,    // [B,L,2*DINNER]: z at DINNER..2*DINNER-1
                 ushort_t*     __restrict__ yb,    // [B,L,DINNER] bf16 out
                 int dir) {
    const int b = blockIdx.x / (DINNER / 256);
    const int d = (blockIdx.x % (DINNER / 256)) * 256 + threadIdx.x;

    float A[DSTATE], h[DSTATE];
#pragma unroll
    for (int s = 0; s < DSTATE; ++s) {
        A[s] = -__expf(Alog[d * DSTATE + s]);
        h[s] = 0.f;
    }
    const float dval = Dp[d];

    __shared__ float sB[DSTATE];
    __shared__ float sC[DSTATE];

    for (int step = 0; step < LL; ++step) {
        const int t = dir ? (LL - 1 - step) : step;
        const size_t row = (size_t)(b * LL + t);

        if (threadIdx.x < DSTATE)
            sB[threadIdx.x] = xdbl[row * XPROJ + DTRANK + threadIdx.x];
        else if (threadIdx.x < 2 * DSTATE)
            sC[threadIdx.x - DSTATE] = xdbl[row * XPROJ + DTRANK + DSTATE + (threadIdx.x - DSTATE)];
        __syncthreads();

        const float dtv = dt[row * DINNER + d];
        const float xv  = bf16_to_f32(xsb[row * DINNER + d]);
        float y = 0.f;
#pragma unroll
        for (int s = 0; s < DSTATE; ++s) {
            float dA = __expf(dtv * A[s]);
            h[s] = dA * h[s] + (dtv * sB[s]) * xv;
            y += h[s] * sC[s];
        }
        const float zv = xz[row * (2 * DINNER) + DINNER + d];
        const float sz = zv / (1.f + __expf(-zv));
        yb[row * DINNER + d] = f32_to_bf16((y + xv * dval) * sz);
        __syncthreads();
    }
}

// ---------------- host-side orchestration ----------------
extern "C" void kernel_launch(void* const* d_in, const int* in_sizes, int n_in,
                              void* d_out, int out_size, void* d_ws, size_t ws_size,
                              hipStream_t stream) {
    (void)in_sizes; (void)n_in; (void)out_size; (void)ws_size;

    const float* x       = (const float*)d_in[0];
    const float* Win[2]  = { (const float*)d_in[1],  (const float*)d_in[10] };
    const float* convw[2]= { (const float*)d_in[2],  (const float*)d_in[11] };
    const float* convb[2]= { (const float*)d_in[3],  (const float*)d_in[12] };
    const float* Wx[2]   = { (const float*)d_in[4],  (const float*)d_in[13] };
    const float* Wdt[2]  = { (const float*)d_in[5],  (const float*)d_in[14] };
    const float* dtb[2]  = { (const float*)d_in[6],  (const float*)d_in[15] };
    const float* Alog[2] = { (const float*)d_in[7],  (const float*)d_in[16] };
    const float* Dp[2]   = { (const float*)d_in[8],  (const float*)d_in[17] };
    const float* Wout[2] = { (const float*)d_in[9],  (const float*)d_in[18] };
    float* out = (float*)d_out;

    // workspace carve-up (256B aligned)
    char* ws = (char*)d_ws;
    size_t off = 0;
    auto carve = [&](size_t bytes) -> char* {
        char* p = ws + off;
        off = (off + bytes + 255) & ~(size_t)255;
        return p;
    };
    ushort_t* winb[2], *wxb[2], *wdtb[2], *woutb[2], *xsb[2], *xdblb[2], *ybuf[2];
    float *xzb[2], *xdbl[2], *dtbuf[2];
    for (int r = 0; r < 2; ++r) winb[r]  = (ushort_t*)carve((size_t)2*DINNER*DMODEL*2);
    for (int r = 0; r < 2; ++r) wxb[r]   = (ushort_t*)carve((size_t)XPROJ*DINNER*2);
    for (int r = 0; r < 2; ++r) wdtb[r]  = (ushort_t*)carve((size_t)DINNER*DTRANK*2);
    for (int r = 0; r < 2; ++r) woutb[r] = (ushort_t*)carve((size_t)DMODEL*DINNER*2);
    ushort_t* xb = (ushort_t*)carve((size_t)MROWS*DMODEL*2);
    for (int r = 0; r < 2; ++r) xzb[r]   = (float*)carve((size_t)MROWS*2*DINNER*4);
    for (int r = 0; r < 2; ++r) xsb[r]   = (ushort_t*)carve((size_t)MROWS*DINNER*2);
    for (int r = 0; r < 2; ++r) xdbl[r]  = (float*)carve((size_t)MROWS*XPROJ*4);
    for (int r = 0; r < 2; ++r) xdblb[r] = (ushort_t*)carve((size_t)MROWS*XPROJ*2);
    for (int r = 0; r < 2; ++r) dtbuf[r] = (float*)carve((size_t)MROWS*DINNER*4);
    for (int r = 0; r < 2; ++r) ybuf[r]  = (ushort_t*)carve((size_t)MROWS*DINNER*2);

    auto cvt = [&](const float* src, ushort_t* dst, int n) {
        cvt_f32_bf16_kernel<<<(n + 255) / 256, 256, 0, stream>>>(src, dst, n);
    };

    // 0) bf16 conversions of weights + input
    for (int r = 0; r < 2; ++r) {
        cvt(Win[r],  winb[r],  2*DINNER*DMODEL);
        cvt(Wx[r],   wxb[r],   XPROJ*DINNER);
        cvt(Wdt[r],  wdtb[r],  DINNER*DTRANK);
        cvt(Wout[r], woutb[r], DMODEL*DINNER);
    }
    cvt(x, xb, MROWS*DMODEL);

    const dim3 wv(32);
    for (int r = 0; r < 2; ++r) {
        // 1) in-projection: xz = x @ Win^T   (M=4096, N=2048, K=512)
        gemm_bf16_wmma_kernel<<<dim3(MROWS/16, (2*DINNER)/64), wv, 0, stream>>>(
            xb, DMODEL, winb[r], DMODEL, xzb[r], 2*DINNER, DMODEL, 0);

        // 2) depthwise conv + SiLU -> xs (bf16)
        conv_silu_kernel<<<(BB*LL*DINNER + 255)/256, 256, 0, stream>>>(
            xzb[r], convw[r], convb[r], xsb[r], r);

        // 3) x-projection: xdbl = xs @ Wx^T  (M=4096, N=64, K=1024)
        gemm_bf16_wmma_kernel<<<dim3(MROWS/16, XPROJ/64), wv, 0, stream>>>(
            xsb[r], DINNER, wxb[r], DINNER, xdbl[r], XPROJ, DINNER, 0);
        cvt(xdbl[r], xdblb[r], MROWS*XPROJ);

        // 4) dt-projection: dt_raw = xdbl[:, :32] @ Wdt^T  (M=4096, N=1024, K=32)
        gemm_bf16_wmma_kernel<<<dim3(MROWS/16, DINNER/64), wv, 0, stream>>>(
            xdblb[r], XPROJ, wdtb[r], DTRANK, dtbuf[r], DINNER, DTRANK, 0);
        softplus_kernel<<<(MROWS*DINNER + 255)/256, 256, 0, stream>>>(
            dtbuf[r], dtb[r], MROWS*DINNER);

        // 5) fused selective scan + Dp skip + z-gate -> y (bf16)
        scan_kernel<<<BB * (DINNER/256), 256, 0, stream>>>(
            dtbuf[r], xsb[r], xdbl[r], Alog[r], Dp[r], xzb[r], ybuf[r], r);

        // 6) out-projection: out (+)= y @ Wout^T  (M=4096, N=512, K=1024)
        gemm_bf16_wmma_kernel<<<dim3(MROWS/16, DMODEL/64), wv, 0, stream>>>(
            ybuf[r], DINNER, woutb[r], DINNER, out, DMODEL, DINNER, /*betaAdd=*/r);
    }
}
